// OneEventExtracter_33036888441068
// MI455X (gfx1250) — matrix operand order repaired and checked by reading, not compiled, not run on hardware
//
#include <hip/hip_runtime.h>

// ---------------------------------------------------------------------------
// OneEventExtracter pointer-logits for MI455X (gfx1250, wave32, WMMA).
// Phase A: reps = embs @ W^T + b  (WMMA bf16, f32 accum), fused RoPE, bf16 ws.
// Phase B: logits[b,h] = q @ k^T  (WMMA bf16), mask, scatter to output layout.
//          q/k tiles staged to LDS with async global->LDS copies
//          (ASYNCcnt-tracked) when the toolchain exposes the gfx1250 builtins.
// ---------------------------------------------------------------------------

typedef __attribute__((ext_vector_type(16))) __bf16 v16bf;
typedef __attribute__((ext_vector_type(8)))  __bf16 v8bf;
typedef __attribute__((ext_vector_type(8)))  float  v8f;
typedef __attribute__((ext_vector_type(4)))  int    v4i;

#if defined(__has_builtin)
#if __has_builtin(__builtin_amdgcn_global_load_async_to_lds_b128) && \
    __has_builtin(__builtin_amdgcn_s_wait_asynccnt)
#define USE_ASYNC_LDS 1
#endif
#endif

#if defined(USE_ASYNC_LDS)
typedef __attribute__((address_space(1))) v4i* gptr_b128;   // global int4*
typedef __attribute__((address_space(3))) v4i* lptr_b128;   // LDS int4*
#endif

namespace cfg {
constexpr int B  = 16;
constexpr int L  = 256;
constexpr int H  = 1024;
constexpr int D  = 64;                    // head dim
constexpr int NH = 62;                    // tri(1) + arg(1) + role(60)
constexpr int NCOL  = 2 * NH * D;         // 7936 projection columns
constexpr int WSCOL = NH * D;             // 3968 cols in each of q_ws / k_ws
constexpr int M  = B * L;                 // 4096 tokens
constexpr size_t TRI_SZ  = (size_t)B * L * L;        // 1,048,576
constexpr size_t ROLE_BASE = 2 * TRI_SZ;             // tri + arg regions
}

// ---- fragment helpers (16-bit A 16x32 / B 32x16 VGPR layouts, wave32) -----
__device__ inline v16bf frag_a(const __bf16* row, int lane) {
    // lane<16: K 0..7 and 16..23 ; lane>=16: K 8..15 and 24..31
    const int sel = (lane >> 4) * 8;
    union { v16bf v; v8bf h[2]; } u;
    u.h[0] = *(const v8bf*)(row + sel);
    u.h[1] = *(const v8bf*)(row + 16 + sel);
    return u.v;
}
__device__ inline v16bf frag_b(const __bf16* row, int lane) {
    // lane<16: K 0..7 and 16..23 ; lane>=16: K 8..15 and 24..31 (row = per-N contiguous K)
    const __bf16* p = row + (lane >> 4) * 8;
    union { v16bf v; v8bf h[2]; } u;
    u.h[0] = *(const v8bf*)(p);
    u.h[1] = *(const v8bf*)(p + 16);
    return u.v;
}

// ---------------------------------------------------------------------------
// Kernel A: projection GEMM + bias + RoPE, writes rotated q/k as bf16.
// Block: 256 threads (8 waves). Tile: M=32, N=64, K-step=32.
// ---------------------------------------------------------------------------
__launch_bounds__(256)
__global__ void proj_rope_kernel(const float* __restrict__ embs,
                                 const float* __restrict__ triW, const float* __restrict__ trib,
                                 const float* __restrict__ argW, const float* __restrict__ argb,
                                 const float* __restrict__ roleW, const float* __restrict__ roleb,
                                 __bf16* __restrict__ qws, __bf16* __restrict__ kws) {
    using namespace cfg;
    constexpr int MB = 32, NB = 64, KB = 32;
    __shared__ __align__(16) __bf16 Alds[MB][KB];   // 2 KB
    __shared__ __align__(16) __bf16 Blds[NB][KB];   // 4 KB
    __shared__ __align__(16) float  Clds[MB][NB];   // 8 KB

    const int t    = threadIdx.x;
    const int lane = t & 31;
    const int wv   = t >> 5;
    const int mBase = blockIdx.y * MB;
    const int nBase = blockIdx.x * NB;
    const int mSub  = (wv >> 2) * 16;
    const int nSub  = (wv & 3) * 16;

    // Per-thread source rows (loop-invariant address math).
    const int arow = t >> 3,  ac4 = (t & 7) * 4;
    const float* asrc = embs + (size_t)(mBase + arow) * H + ac4;

    const float* bsrc[2];
    int brow[2], bc4[2];
    #pragma unroll
    for (int hblk = 0; hblk < 2; ++hblk) {
        const int task = t + hblk * 256;
        brow[hblk] = task >> 3;  bc4[hblk] = (task & 7) * 4;
        const int c = nBase + brow[hblk];
        const float* wp;
        if (c < 128)      wp = triW  + (size_t)c * H;
        else if (c < 256) wp = argW  + (size_t)(c - 128) * H;
        else              wp = roleW + (size_t)(c - 256) * H;
        bsrc[hblk] = wp + bc4[hblk];
    }

    v8f acc = {};

    for (int kb = 0; kb < H; kb += KB) {
        // Prefetch next K-step tiles (lowers to global_prefetch_b8).
        if (kb + KB < H) {
            __builtin_prefetch(asrc + kb + KB, 0, 1);
            __builtin_prefetch(bsrc[0] + kb + KB, 0, 1);
            __builtin_prefetch(bsrc[1] + kb + KB, 0, 1);
        }
        // A tile: 32x32 f32 -> bf16 (one float4 per thread)
        {
            const float4 v = *(const float4*)(asrc + kb);
            Alds[arow][ac4 + 0] = (__bf16)v.x;  Alds[arow][ac4 + 1] = (__bf16)v.y;
            Alds[arow][ac4 + 2] = (__bf16)v.z;  Alds[arow][ac4 + 3] = (__bf16)v.w;
        }
        // B tile: 64 weight rows x 32 (two float4 per thread); row-major W is
        // already the per-N-contiguous-K layout the B fragment wants.
        #pragma unroll
        for (int hblk = 0; hblk < 2; ++hblk) {
            const float4 v = *(const float4*)(bsrc[hblk] + kb);
            const int row = brow[hblk], c4 = bc4[hblk];
            Blds[row][c4 + 0] = (__bf16)v.x;  Blds[row][c4 + 1] = (__bf16)v.y;
            Blds[row][c4 + 2] = (__bf16)v.z;  Blds[row][c4 + 3] = (__bf16)v.w;
        }
        __syncthreads();

        const v16bf a = frag_a(&Alds[mSub + (lane & 15)][0], lane);
        const v16bf b = frag_b(&Blds[nSub + (lane & 15)][0], lane);
        acc = __builtin_amdgcn_wmma_f32_16x16x32_bf16(false, a, false, b,
                                                      (short)0, acc, false, false);
        __syncthreads();
    }

    // Accumulators -> LDS (C layout: VGPR r, lane l -> M = r + 8*(l>=16), N = l%16)
    {
        const int m = mSub + (lane >> 4) * 8;
        const int n = nSub + (lane & 15);
        #pragma unroll
        for (int r = 0; r < 8; ++r) Clds[m + r][n] = acc[r];
    }
    __syncthreads();

    // Bias + RoPE epilogue. Thread handles row (t>>3), 8 cols (t&7)*8..+7.
    // Head halves are 64-aligned, so an 8-col chunk never straddles q/k or a
    // rotation pair; rotation pairs (2i, 2i+1) are local to the chunk.
    {
        const int ml = t >> 3;
        const int c0 = (t & 7) * 8;
        const int mg = mBase + ml;
        const int pos = mg & (L - 1);
        const int cg0 = nBase + c0;

        __align__(16) __bf16 outv[8];
        #pragma unroll
        for (int j = 0; j < 8; j += 2) {
            const int c = cg0 + j;
            float be, bo; int d;
            if (c < 128)      { d = c & 63;                 be = trib[c];        bo = trib[c + 1]; }
            else if (c < 256) { const int lc = c - 128; d = lc & 63; be = argb[lc]; bo = argb[lc + 1]; }
            else              { const int c2 = c - 256; d = c2 & 63; be = roleb[c2]; bo = roleb[c2 + 1]; }
            const float xe = Clds[ml][c0 + j]     + be;
            const float xo = Clds[ml][c0 + j + 1] + bo;
            const float fr  = __powf(10000.0f, -(float)d * (1.0f / 64.0f)); // d even
            const float ang = (float)pos * fr;
            const float cs = __cosf(ang), sn = __sinf(ang);
            outv[j]     = (__bf16)(xe * cs - xo * sn);
            outv[j + 1] = (__bf16)(xo * cs + xe * sn);
        }
        // destination column (contiguous 8): head*64 + d0
        int head, half, d0;
        if (cg0 < 128)      { head = 0;             half = cg0 >> 6;         d0 = cg0 & 63; }
        else if (cg0 < 256) { const int lc = cg0 - 128; head = 1; half = lc >> 6; d0 = lc & 63; }
        else { const int c2 = cg0 - 256; head = 2 + (c2 >> 7); const int lc = c2 & 127;
               half = lc >> 6; d0 = lc & 63; }
        __bf16* dst = (half ? kws : qws) + (size_t)mg * WSCOL + head * D + d0;
        *(uint4*)dst = *(const uint4*)outv;
    }
}

// ---------------------------------------------------------------------------
// Kernel B: per (batch, head) attention logits q @ k^T, masked, scattered out.
// Block: 256 threads (8 waves), tile 64x64; each wave: 2 m-reps of 16x16,
// K=64 as two 16x16x32 WMMAs per tile.
// ---------------------------------------------------------------------------
__launch_bounds__(256)
__global__ void attn_kernel(const __bf16* __restrict__ qws, const __bf16* __restrict__ kws,
                            const unsigned char* __restrict__ wordMask,
                            const unsigned char* __restrict__ triuMask,
                            float* __restrict__ out) {
    using namespace cfg;
    constexpr int MBB = 64, NBB = 64;
    __shared__ __align__(16) __bf16 Qlds[MBB][D];   // 8 KB
    __shared__ __align__(16) __bf16 Klds[NBB][D];   // 8 KB

    const int t    = threadIdx.x;
    const int lane = t & 31;
    const int wv   = t >> 5;
    const int bh   = blockIdx.z;
    const int b    = bh / NH;
    const int head = bh - b * NH;
    const int mBase = blockIdx.y * MBB;
    const int nBase = blockIdx.x * NBB;
    const size_t hcol = (size_t)head * D;

    // Stage q (64x64) and k (64x64) bf16 tiles (16B per thread per chunk).
    #pragma unroll
    for (int hblk = 0; hblk < 2; ++hblk) {
        const int task = t + hblk * 256;
        const int row = task >> 3, c8 = (task & 7) * 8;
        const __bf16* qsrc = qws + (size_t)(b * L + mBase + row) * WSCOL + hcol + c8;
        const __bf16* ksrc = kws + (size_t)(b * L + nBase + row) * WSCOL + hcol + c8;
#if defined(USE_ASYNC_LDS)
        // ASYNCcnt-tracked direct global->LDS copy (no VGPR round-trip).
        __builtin_amdgcn_global_load_async_to_lds_b128(
            (gptr_b128)qsrc, (lptr_b128)&Qlds[row][c8], 0, 0);
        __builtin_amdgcn_global_load_async_to_lds_b128(
            (gptr_b128)ksrc, (lptr_b128)&Klds[row][c8], 0, 0);
#else
        *(uint4*)&Qlds[row][c8] = *(const uint4*)qsrc;
        *(uint4*)&Klds[row][c8] = *(const uint4*)ksrc;
#endif
    }
#if defined(USE_ASYNC_LDS)
    __builtin_amdgcn_s_wait_asynccnt(0);
#endif
    __syncthreads();

    const int nSub  = (wv & 3) * 16;
    const int mSub0 = (wv >> 2) * 32;

    v8f acc0 = {}, acc1 = {};
    #pragma unroll
    for (int kc = 0; kc < 2; ++kc) {
        const int ks = kc * 32;
        const v16bf bfr = frag_b(&Klds[nSub + (lane & 15)][ks], lane);
        const v16bf a0  = frag_a(&Qlds[mSub0 +      (lane & 15)][ks], lane);
        acc0 = __builtin_amdgcn_wmma_f32_16x16x32_bf16(false, a0, false, bfr,
                                                       (short)0, acc0, false, false);
        const v16bf a1  = frag_a(&Qlds[mSub0 + 16 + (lane & 15)][ks], lane);
        acc1 = __builtin_amdgcn_wmma_f32_16x16x32_bf16(false, a1, false, bfr,
                                                       (short)0, acc1, false, false);
    }

    // Masked scatter to the three output regions.
    const int n = nBase + nSub + (lane & 15);
    const unsigned char* mask = (head >= 2) ? triuMask : wordMask;

    #pragma unroll
    for (int rep = 0; rep < 2; ++rep) {
        const v8f acc = rep ? acc1 : acc0;
        const int mTop = mBase + mSub0 + rep * 16 + (lane >> 4) * 8;
        #pragma unroll
        for (int r = 0; r < 8; ++r) {
            const int m = mTop + r;
            const size_t mn = ((size_t)b * L + m) * L + n;
            const float v = mask[mn] ? acc[r] : -1.0e12f;
            size_t idx;
            if (head == 0)      idx = mn;                                   // tri  [B,L,L,1]
            else if (head == 1) idx = TRI_SZ + mn;                          // arg  [B,L,L,1]
            else                idx = ROLE_BASE + mn * 60 + (head - 2);     // role [B,L,L,1,60]
            out[idx] = v;
        }
    }
}

// ---------------------------------------------------------------------------
extern "C" void kernel_launch(void* const* d_in, const int* in_sizes, int n_in,
                              void* d_out, int out_size, void* d_ws, size_t ws_size,
                              hipStream_t stream) {
    using namespace cfg;
    const float* embs  = (const float*)d_in[0];
    const float* triW  = (const float*)d_in[1];
    const float* trib  = (const float*)d_in[2];
    const float* argW  = (const float*)d_in[3];
    const float* argb  = (const float*)d_in[4];
    const float* roleW = (const float*)d_in[5];
    const float* roleb = (const float*)d_in[6];
    const unsigned char* wmask = (const unsigned char*)d_in[7];
    const unsigned char* tmask = (const unsigned char*)d_in[8];

    __bf16* qws = (__bf16*)d_ws;                         // [4096][3968] bf16
    __bf16* kws = qws + (size_t)M * WSCOL;               // [4096][3968] bf16

    dim3 gA(NCOL / 64, M / 32);                          // (124, 128)
    proj_rope_kernel<<<gA, 256, 0, stream>>>(embs, triW, trib, argW, argb,
                                             roleW, roleb, qws, kws);

    dim3 gB(L / 64, L / 64, B * NH);                     // (4, 4, 992)
    attn_kernel<<<gB, 256, 0, stream>>>(qws, kws, wmask, tmask, (float*)d_out);
}
